// SimpleMambaImplementation_18777597018878
// MI455X (gfx1250) — matrix operand order repaired and checked by reading, not compiled, not run on hardware
//
#include <hip/hip_runtime.h>
#include <stdint.h>

#define B_   2
#define L_   2048
#define DM_  512
#define DS_  16
#define DI_  1024
#define M_   (B_ * L_)     // 4096 rows of activations
#define NIN_ (2 * DI_)     // 2048 in_proj outputs

typedef __attribute__((ext_vector_type(16))) __bf16 v16bf;
typedef __attribute__((ext_vector_type(8)))  float  v8f;
typedef __attribute__((ext_vector_type(4)))  unsigned int v4u;
typedef __attribute__((ext_vector_type(8)))  unsigned int v8u;

union ABFrag { v16bf v; uint32_t u[8]; };

__device__ __forceinline__ unsigned short f2bf(float f) {
  union { float f; uint32_t u; } x; x.f = f;
  uint32_t r = x.u + 0x7FFFu + ((x.u >> 16) & 1u);   // round-to-nearest-even
  return (unsigned short)(r >> 16);
}

// ---------------------------------------------------------------- cvt f32->bf16
__global__ void cvt_bf16_kernel(const float* __restrict__ in,
                                unsigned short* __restrict__ out, int n) {
  int i = blockIdx.x * blockDim.x + threadIdx.x;
  if (i < n) out[i] = f2bf(in[i]);
}

// ---------------------------------------------------------------- TDM staging
// Stage a 16xK bf16 A-tile (row-major, row stride K) into LDS via the Tensor
// Data Mover.  D# per ISA 8.3/8.4: group0 = {count, lds_addr, global_addr,
// type=2}; group1 = {data_size=2B, pad 1 dword per 256 dwords, dims/strides}.
// The pad gives the 16 LDS rows a 257/514-dword pitch (bank-conflict-free
// fragment reads); readers must index with t + (t>>8).
__device__ __forceinline__ void stage_a_tile_tdm(const unsigned short* gA, int K,
                                                 int row0, uint32_t* sA) {
  uint64_t gaddr = (uint64_t)(uintptr_t)(gA + (size_t)row0 * K);
  uint32_t laddr = (uint32_t)(uintptr_t)sA;        // flat LDS addr[31:0] = offset
  v4u g0;
  g0.x = 1u;                                        // count=1, load, user mode
  g0.y = laddr;                                     // lds_addr
  g0.z = (uint32_t)gaddr;                           // global_addr[31:0]
  g0.w = (uint32_t)((gaddr >> 32) & 0x01FFFFFFu) | (2u << 30);  // addr[56:32]|type=2
  v8u g1;
  g1.s0 = (1u << 16) | (1u << 20) | (7u << 22);     // data_size=2B, pad_en, ivl=256dw, amt=1dw
  g1.s1 = ((uint32_t)K & 0xFFFFu) << 16;            // tensor_dim0[15:0]
  g1.s2 = (16u << 16);                              // tensor_dim0[31:16]=0 | tensor_dim1=16
  g1.s3 = ((uint32_t)K) << 16;                      // tensor_dim1[31:16]=0 | tile_dim0=K
  g1.s4 = 16u;                                      // tile_dim1=16, tile_dim2=0
  g1.s5 = (uint32_t)K;                              // tensor_dim0_stride[31:0]
  g1.s6 = 0u;                                       // stride[47:32], dim1_stride[15:0]
  g1.s7 = 0u;
  if (threadIdx.x == 0) {
    asm volatile("tensor_load_to_lds %0, %1" :: "s"(g0), "s"(g1) : "memory");
    __builtin_amdgcn_s_wait_tensorcnt(0);
  }
  __syncthreads();
}

// ---------------------------------------------------------------- WMMA fragments
// A-fragment from the padded LDS tile.  ISA 7.12.2 (16-bit A 16x32): lane
// m = lane&15; lanes 0-15 hold K 0..7 (v0-3) / 16..23 (v4-7); lanes 16-31
// hold K 8..15 / 24..31; 2 bf16 per VGPR.
__device__ __forceinline__ void load_a_frag_lds(const uint32_t* sA, int Kd,
                                                int k0, ABFrag& a) {
  int lane = threadIdx.x & 31;
  int m = lane & 15, half = lane >> 4;
  #pragma unroll
  for (int v = 0; v < 8; ++v) {
    int bk = (v < 4) ? (2 * v + 8 * half) : (16 + 2 * (v - 4) + 8 * half);
    int t = m * Kd + ((k0 + bk) >> 1);              // dword index before pad
    a.u[v] = sA[t + (t >> 8)];                      // TDM pad skew
  }
}

// B-fragment: 32x16 tile of W^T, W is [N, K] row-major (weights, from global).
// Lane n = lane&15; lanes 0-15 hold K 0..7 (v0-3) / 16..23 (v4-7); lanes 16-31
// hold K 8..15 / 24..31; 2 bf16 per VGPR.
__device__ __forceinline__ void load_b_frag(const unsigned short* W, int ldw,
                                            int n0, int k0, ABFrag& b) {
  int lane = threadIdx.x & 31;
  int n = lane & 15, half = lane >> 4;
  const uint32_t* p = reinterpret_cast<const uint32_t*>(
      W + (size_t)(n0 + n) * ldw + k0);
  #pragma unroll
  for (int v = 0; v < 8; ++v) {
    b.u[v] = p[((v < 4) ? (2 * v + 8 * half) : (16 + 2 * (v - 4) + 8 * half)) >> 1];
  }
}

// 4 N-tiles per wave: one A-fragment feeds 4 WMMAs per k-step (4x A reuse).
template <int K>
__device__ __forceinline__ void wmma_wave_4tiles(const uint32_t* sA,
                                                 const unsigned short* W,
                                                 int n0, v8f acc[4]) {
  for (int k0 = 0; k0 < K; k0 += 32) {
    ABFrag a;
    load_a_frag_lds(sA, K / 2, k0, a);
    #pragma unroll
    for (int j = 0; j < 4; ++j) {
      ABFrag b;
      load_b_frag(W, K, n0 + j * 16, k0, b);
      acc[j] = __builtin_amdgcn_wmma_f32_16x16x32_bf16(
          false, a.v, false, b.v, (short)0, acc[j], false, false);
    }
  }
}

// ---------------------------------------------------------------- GEMM kernels
// Block = 256 threads (8 waves); block covers 16 rows x 512 cols.
// xz = x @ W_in^T, split into xi (cols < 1024) and z.
__global__ void __launch_bounds__(256)
gemm_xz_kernel(const unsigned short* __restrict__ xbf,
               const unsigned short* __restrict__ Wbf,
               float* __restrict__ xi, float* __restrict__ z) {
  extern __shared__ uint32_t sA[];
  int row0 = blockIdx.x * 16;
  stage_a_tile_tdm(xbf, DM_, row0, sA);
  int wave = threadIdx.x >> 5;
  int n0 = blockIdx.y * 512 + wave * 64;
  v8f acc[4] = {};
  wmma_wave_4tiles<DM_>(sA, Wbf, n0, acc);
  int lane = threadIdx.x & 31;
  int n = lane & 15, half = lane >> 4;
  #pragma unroll
  for (int j = 0; j < 4; ++j) {
    int col = n0 + j * 16 + n;
    float* dst = (col < DI_) ? (xi + col) : (z + (col - DI_));
    #pragma unroll
    for (int r = 0; r < 8; ++r) {
      int m = r + 8 * half;                         // C/D layout: vgpr r, half sel
      dst[(size_t)(row0 + m) * DI_] = acc[j][r];
    }
  }
}

// delta = softplus(h @ W_dt^T + b_dt)
__global__ void __launch_bounds__(256)
gemm_dt_kernel(const unsigned short* __restrict__ hbf,
               const unsigned short* __restrict__ Wbf,
               const float* __restrict__ b_dt, float* __restrict__ delta) {
  extern __shared__ uint32_t sA[];
  int row0 = blockIdx.x * 16;
  stage_a_tile_tdm(hbf, DI_, row0, sA);
  int wave = threadIdx.x >> 5;
  int n0 = blockIdx.y * 512 + wave * 64;
  v8f acc[4] = {};
  wmma_wave_4tiles<DI_>(sA, Wbf, n0, acc);
  int lane = threadIdx.x & 31;
  int n = lane & 15, half = lane >> 4;
  #pragma unroll
  for (int j = 0; j < 4; ++j) {
    int col = n0 + j * 16 + n;
    float bias = b_dt[col];
    #pragma unroll
    for (int r = 0; r < 8; ++r) {
      int m = r + 8 * half;
      float v = acc[j][r] + bias;
      float sp = (v > 20.f) ? v : __logf(1.f + __expf(v));   // softplus
      delta[(size_t)(row0 + m) * DI_ + col] = sp;
    }
  }
}

// out = y_gated @ W_out^T
__global__ void __launch_bounds__(256)
gemm_out_kernel(const unsigned short* __restrict__ ybf,
                const unsigned short* __restrict__ Wbf,
                float* __restrict__ out) {
  extern __shared__ uint32_t sA[];
  int row0 = blockIdx.x * 16;
  stage_a_tile_tdm(ybf, DI_, row0, sA);
  int wave = threadIdx.x >> 5;
  int n0 = wave * 64;                               // N = 512 per block exactly
  v8f acc[4] = {};
  wmma_wave_4tiles<DI_>(sA, Wbf, n0, acc);
  int lane = threadIdx.x & 31;
  int n = lane & 15, half = lane >> 4;
  #pragma unroll
  for (int j = 0; j < 4; ++j) {
    int col = n0 + j * 16 + n;
    #pragma unroll
    for (int r = 0; r < 8; ++r) {
      int m = r + 8 * half;
      out[(size_t)(row0 + m) * DM_ + col] = acc[j][r];
    }
  }
}

// ---------------------------------------------------------------- conv + SiLU
__global__ void conv_silu_kernel(const float* __restrict__ xi,
                                 const float* __restrict__ cw,
                                 const float* __restrict__ cb,
                                 float* __restrict__ h,
                                 unsigned short* __restrict__ hbf) {
  int i = blockIdx.x * blockDim.x + threadIdx.x;
  if (i >= B_ * L_ * DI_) return;
  int c = i % DI_;
  int l = (i / DI_) % L_;
  int b = i / (DI_ * L_);
  const float* base = xi + (size_t)b * L_ * DI_ + c;
  float acc = cb[c];
  #pragma unroll
  for (int j = 0; j < 4; ++j) {                     // causal left-pad of 3
    int ll = l - 3 + j;
    if (ll >= 0) acc += cw[c * 4 + j] * base[(size_t)ll * DI_];
  }
  float s = acc / (1.f + __expf(-acc));             // silu
  h[i] = s;
  hbf[i] = f2bf(s);
}

// ---------------------------------------------------------------- B-projection
// Bmat[row, s] = h[row, :] . W_x[16+s, :]   (second half rows of W_x)
__global__ void bmat_kernel(const float* __restrict__ h,
                            const float* __restrict__ W_x,
                            float* __restrict__ Bm) {
  int t = blockIdx.x * blockDim.x + threadIdx.x;
  if (t >= M_ * DS_) return;
  int s = t & (DS_ - 1);
  int row = t >> 4;
  const float* wr = W_x + (size_t)(DS_ + s) * DI_;
  const float* hr = h + (size_t)row * DI_;
  float acc = 0.f;
  for (int k = 0; k < DI_; ++k) acc += hr[k] * wr[k];
  Bm[t] = acc;
}

// ---------------------------------------------------------------- selective scan
// One lane per (b, c, s); 16-lane groups reduce y across the state dim.
__global__ void __launch_bounds__(256)
scan_kernel(const float* __restrict__ delta, const float* __restrict__ Bm,
            const float* __restrict__ h, const float* __restrict__ A_log,
            float* __restrict__ y) {
  int t = blockIdx.x * blockDim.x + threadIdx.x;    // 0 .. B_*DI_*DS_-1
  int s = t & (DS_ - 1);
  int c = (t >> 4) & (DI_ - 1);
  int b = t >> 14;
  float A = -__expf(A_log[c * DS_ + s]);
  float state = 0.f;
  const float* dptr = delta + (size_t)b * L_ * DI_ + c;
  const float* hptr = h     + (size_t)b * L_ * DI_ + c;
  const float* Bptr = Bm    + (size_t)b * L_ * DS_ + s;
  float* yptr       = y     + (size_t)b * L_ * DI_ + c;
  for (int l = 0; l < L_; ++l) {
    if (l + 16 < L_) {                              // hide strided-load latency
      __builtin_prefetch(dptr + (size_t)(l + 16) * DI_, 0, 1);
      __builtin_prefetch(hptr + (size_t)(l + 16) * DI_, 0, 1);
    }
    float d  = dptr[(size_t)l * DI_];
    float hv = hptr[(size_t)l * DI_];
    float Bv = Bptr[(size_t)l * DS_];
    float dA = __expf(d * A);
    state = dA * state + d * Bv * hv;
    float partial = state * Bv;                     // C == B in the reference
    partial += __shfl_xor(partial, 1, 32);
    partial += __shfl_xor(partial, 2, 32);
    partial += __shfl_xor(partial, 4, 32);
    partial += __shfl_xor(partial, 8, 32);
    if (s == 0) yptr[(size_t)l * DI_] = partial;
  }
}

// ---------------------------------------------------------------- gate
__global__ void gate_kernel(const float* __restrict__ y, const float* __restrict__ h,
                            const float* __restrict__ z, const float* __restrict__ Dp,
                            unsigned short* __restrict__ ybf) {
  int i = blockIdx.x * blockDim.x + threadIdx.x;
  if (i >= B_ * L_ * DI_) return;
  int c = i % DI_;
  float zz = z[i];
  float g = zz / (1.f + __expf(-zz));               // silu(z)
  ybf[i] = f2bf((y[i] + Dp[c] * h[i]) * g);
}

// ---------------------------------------------------------------- launch
static inline size_t smem_bytes(int K) {            // 16 padded LDS rows
  int kd = K / 2;
  return (size_t)16 * (kd + kd / 256) * 4;
}

extern "C" void kernel_launch(void* const* d_in, const int* in_sizes, int n_in,
                              void* d_out, int out_size, void* d_ws, size_t ws_size,
                              hipStream_t stream) {
  const float* x      = (const float*)d_in[0];
  const float* W_in   = (const float*)d_in[1];
  const float* conv_w = (const float*)d_in[2];
  const float* conv_b = (const float*)d_in[3];
  const float* W_x    = (const float*)d_in[4];
  const float* W_dt   = (const float*)d_in[5];
  const float* b_dt   = (const float*)d_in[6];
  const float* A_log  = (const float*)d_in[7];
  const float* Dp     = (const float*)d_in[8];
  const float* W_out  = (const float*)d_in[9];
  float* out = (float*)d_out;
  char* ws = (char*)d_ws;

  size_t off = 0;
  auto alloc = [&](size_t bytes) {
    size_t o = off;
    off += (bytes + 255) & ~(size_t)255;
    return o;
  };
  unsigned short* xbf    = (unsigned short*)(ws + alloc((size_t)M_ * DM_ * 2));
  unsigned short* Winbf  = (unsigned short*)(ws + alloc((size_t)NIN_ * DM_ * 2));
  unsigned short* Wdtbf  = (unsigned short*)(ws + alloc((size_t)DI_ * DI_ * 2));
  unsigned short* Woutbf = (unsigned short*)(ws + alloc((size_t)DM_ * DI_ * 2));
  float* xi   = (float*)(ws + alloc((size_t)M_ * DI_ * 4));
  float* z    = (float*)(ws + alloc((size_t)M_ * DI_ * 4));
  float* h    = (float*)(ws + alloc((size_t)M_ * DI_ * 4));
  unsigned short* hbf = (unsigned short*)(ws + alloc((size_t)M_ * DI_ * 2));
  float* Bm   = (float*)(ws + alloc((size_t)M_ * DS_ * 4));
  float* y    = (float*)(ws + alloc((size_t)M_ * DI_ * 4));
  unsigned short* ybf = (unsigned short*)(ws + alloc((size_t)M_ * DI_ * 2));
  float* delta = xi;   // alias: xi is dead once conv_silu has produced h

  const int T = 256;
  // 1. bf16 copies of activations/weights for WMMA
  cvt_bf16_kernel<<<(M_ * DM_ + T - 1) / T, T, 0, stream>>>(x, xbf, M_ * DM_);
  cvt_bf16_kernel<<<(NIN_ * DM_ + T - 1) / T, T, 0, stream>>>(W_in, Winbf, NIN_ * DM_);
  cvt_bf16_kernel<<<(DI_ * DI_ + T - 1) / T, T, 0, stream>>>(W_dt, Wdtbf, DI_ * DI_);
  cvt_bf16_kernel<<<(DM_ * DI_ + T - 1) / T, T, 0, stream>>>(W_out, Woutbf, DM_ * DI_);
  // 2. in_proj GEMM (M=4096, N=2048, K=512), TDM-staged A, split xi/z
  gemm_xz_kernel<<<dim3(M_ / 16, NIN_ / 512), T, smem_bytes(DM_), stream>>>(
      xbf, Winbf, xi, z);
  // 3. depthwise causal conv + SiLU -> h (f32 + bf16)
  conv_silu_kernel<<<(M_ * DI_ + T - 1) / T, T, 0, stream>>>(xi, conv_w, conv_b, h, hbf);
  // 4. B projection (N=16, VALU dot products)
  bmat_kernel<<<(M_ * DS_ + T - 1) / T, T, 0, stream>>>(h, W_x, Bm);
  // 5. dt GEMM (M=4096, N=1024, K=1024) + softplus -> delta (overwrites xi)
  gemm_dt_kernel<<<dim3(M_ / 16, DI_ / 512), T, smem_bytes(DI_), stream>>>(
      hbf, Wdtbf, b_dt, delta);
  // 6. selective scan (serial over L, 32K lanes)
  scan_kernel<<<(B_ * DI_ * DS_) / T, T, 0, stream>>>(delta, Bm, h, A_log, y);
  // 7. gate: (y + D*h) * silu(z) -> bf16
  gate_kernel<<<(M_ * DI_ + T - 1) / T, T, 0, stream>>>(y, h, z, Dp, ybf);
  // 8. out_proj GEMM (M=4096, N=512, K=1024)
  gemm_out_kernel<<<dim3(M_ / 16, 1), T, smem_bytes(DI_), stream>>>(ybf, Woutbf, out);
}
